// AdaptiveTopologicalAttention_3229815407294
// MI455X (gfx1250) — compile-verified, matrix operand order checked by reading, not run on hardware
//
#include <hip/hip_runtime.h>
#include <hip/hip_bf16.h>
#include <stdint.h>

#define D_MODEL 256
#define HD      64
#define BB      4
#define NN      2048
#define EE      65536
#define KEEP    1024   // k = N * (1 - 0.5)

typedef __attribute__((ext_vector_type(16))) __bf16 v16bf;
typedef __attribute__((ext_vector_type(8)))  float  v8f;

union Frag16 { uint4 u[2]; v16bf v; };

__device__ __forceinline__ unsigned short f2bf(float f) {
    __hip_bfloat16 h = __float2bfloat16(f);     // RNE, hardware cvt path
    return __builtin_bit_cast(unsigned short, h);
}

// ---------------------------------------------------------------- fp32 -> bf16 bulk convert (vectorized)
__global__ void k_cvt_bf16(const float* __restrict__ src, unsigned short* __restrict__ dst, int n4) {
    int t = blockIdx.x * 256 + threadIdx.x;
    if (t >= n4) return;
    float4 v = ((const float4*)src)[t];
    unsigned d0 = f2bf(v.x), d1 = f2bf(v.y), d2 = f2bf(v.z), d3 = f2bf(v.w);
    uint2 o; o.x = d0 | (d1 << 16); o.y = d2 | (d3 << 16);
    ((uint2*)dst)[t] = o;
}

// ---------------------------------------------------------------- weight convert + transpose: WT[n][k] = bf16(W[k][n])
__global__ void k_cvt_w_t(const float* __restrict__ W, unsigned short* __restrict__ WT) {
    int t = blockIdx.x * 256 + threadIdx.x;     // 65536
    int n = t & 255, k = t >> 8;
    WT[(size_t)n * D_MODEL + k] = f2bf(W[(size_t)k * D_MODEL + n]);
}

// ---------------------------------------------------------------- edge scatter (row-major + transposed)
__global__ void k_scatter_edges(const long long* __restrict__ ei,
                                unsigned char* __restrict__ edge,
                                unsigned char* __restrict__ edgeT) {
    int e = blockIdx.x * 256 + threadIdx.x;
    if (e < EE) {
        int r = (int)ei[e];
        int c = (int)ei[EE + e];
        edge [r * NN + c] = 1;
        edgeT[c * NN + r] = 1;
    }
}

// ---------------------------------------------------------------- GNN aggregate
__global__ void k_gnn_aggregate(const float* __restrict__ x,
                                const long long* __restrict__ ei,
                                float* __restrict__ aggr) {
    int t = blockIdx.x * 256 + threadIdx.x;           // BB*EE*64 = 16.7M
    int b   = t / (EE * 64);
    int rem = t % (EE * 64);
    int e   = rem >> 6;
    int c   = rem & 63;
    int row = (int)ei[e];
    int col = (int)ei[EE + e];
    const float4 xv = *(const float4*)(x + ((size_t)(b * NN + col)) * D_MODEL + c * 4);
    float* dst = aggr + ((size_t)(b * NN + row)) * D_MODEL + c * 4;
    atomicAdd(dst + 0, xv.x);
    atomicAdd(dst + 1, xv.y);
    atomicAdd(dst + 2, xv.z);
    atomicAdd(dst + 3, xv.w);
}

// ---------------------------------------------------------------- GNN MLP -> topo score
__global__ __launch_bounds__(128)
void k_gnn_mlp(const float* __restrict__ aggr, const float* __restrict__ Wg1,
               const float* __restrict__ bg1, const float* __restrict__ Wg2,
               const float* __restrict__ bg2, float* __restrict__ topo) {
    int node = blockIdx.x;             // b*NN + n
    int h = threadIdx.x;               // 0..127
    const float* a = aggr + (size_t)node * D_MODEL;
    float s = bg1[h];
    for (int k = 0; k < D_MODEL; ++k) s += a[k] * Wg1[k * 128 + h];
    s = fmaxf(s, 0.f) * Wg2[h];
    __shared__ float red[128];
    red[h] = s; __syncthreads();
    for (int off = 64; off > 0; off >>= 1) {
        if (h < off) red[h] += red[h + off];
        __syncthreads();
    }
    if (h == 0) topo[node] = red[0] + bg2[0];
}

// ---------------------------------------------------------------- top-k column mask (stable rank)
__global__ void k_topk(const float* __restrict__ topo, float* __restrict__ colmask) {
    int t = blockIdx.x * 256 + threadIdx.x;
    if (t >= BB * NN) return;
    int b = t / NN, i = t % NN;
    const float* s = topo + b * NN;
    float si = s[i];
    int rank = 0;
    for (int j = 0; j < NN; ++j) {
        float sj = s[j];
        rank += (sj > si) || (sj == si && j < i);
    }
    colmask[t] = (rank < KEEP) ? 1.f : 0.f;
}

// ---------------------------------------------------------------- WMMA GEMM (all-bf16, no LDS):
// C[M,256] = A[M,256] * W + bias, with W given pre-transposed WT[n][k].
// Each wave: 16 rows x 64 cols (4 accumulators, A-fragment reused 4x per k-step).
// mode 0: bf16 row-major out; mode 1: bf16 transposed [b][h][d][n]; mode 2: fp32 row-major out
__global__ __launch_bounds__(128)
void k_gemm256(const unsigned short* __restrict__ A, const unsigned short* __restrict__ WT,
               const float* __restrict__ bias, void* __restrict__ out, int mode) {
    int w    = threadIdx.x >> 5;
    int lane = threadIdx.x & 31;
    int ln   = lane & 15;
    int half = lane >> 4;
    int m0 = (blockIdx.x * 4 + w) * 16;
    int n0 = blockIdx.y * 64;
    v8f acc[4] = {};
    const unsigned short* arow = A + (size_t)(m0 + ln) * D_MODEL;
    for (int kk = 0; kk < 8; ++kk) {
        int k0 = kk * 32;
        Frag16 af;
        af.u[0] = *(const uint4*)(arow + k0 + half * 8);
        af.u[1] = *(const uint4*)(arow + k0 + 16 + half * 8);
        for (int c = 0; c < 4; ++c) {
            const unsigned short* wrow = WT + (size_t)(n0 + c * 16 + ln) * D_MODEL + k0;
            Frag16 bfrag;
            bfrag.u[0] = *(const uint4*)(wrow + half * 8);
            bfrag.u[1] = *(const uint4*)(wrow + 16 + half * 8);
            acc[c] = __builtin_amdgcn_wmma_f32_16x16x32_bf16(false, af.v, false, bfrag.v,
                                                             (short)0, acc[c], false, false);
        }
    }
    for (int c = 0; c < 4; ++c) {
        int col = n0 + c * 16 + ln;
        float bv = bias[col];
        for (int r = 0; r < 8; ++r) {
            int mg = m0 + r + 8 * half;
            float v = acc[c][r] + bv;
            if (mode == 0) {
                ((unsigned short*)out)[(size_t)mg * D_MODEL + col] = f2bf(v);
            } else if (mode == 1) {
                int b = mg >> 11, n = mg & 2047, hh = col >> 6, dd = col & 63;
                ((unsigned short*)out)[((size_t)((b * 4 + hh) * HD + dd)) * NN + n] = f2bf(v);
            } else {
                ((float*)out)[(size_t)mg * D_MODEL + col] = v;
            }
        }
    }
}

// ---------------------------------------------------------------- two-pass flash attention with mask/renorm
__global__ __launch_bounds__(256)
void k_attention(const unsigned short* __restrict__ Qb, const unsigned short* __restrict__ Kb,
                 const unsigned short* __restrict__ Vt, const unsigned char* __restrict__ edgeT,
                 const float* __restrict__ colmask, unsigned short* __restrict__ attnb) {
    __shared__ unsigned short Pst[8][16][40];       // per-wave P staging (C-layout -> A-layout)
    int wid  = threadIdx.x >> 5;
    int lane = threadIdx.x & 31;
    int ln   = lane & 15;
    int half = lane >> 4;
    int gw = blockIdx.x * 8 + wid;                  // 2048 waves total
    int qt = gw & 127;
    int h  = (gw >> 7) & 3;
    int b  = gw >> 9;
    int q0 = qt * 16;
    const float scale = 0.125f;                     // 1/sqrt(64)

    Frag16 qf0, qf1;                                // Q rows q0+ln, dims 0..63
    const unsigned short* qrow = Qb + ((size_t)(b * NN + q0 + ln)) * D_MODEL + h * HD;
    qf0.u[0] = *(const uint4*)(qrow + half * 8);
    qf0.u[1] = *(const uint4*)(qrow + 16 + half * 8);
    qf1.u[0] = *(const uint4*)(qrow + 32 + half * 8);
    qf1.u[1] = *(const uint4*)(qrow + 48 + half * 8);

    const unsigned short* Kbase = Kb + ((size_t)(b * NN)) * D_MODEL + h * HD;
    const unsigned short* Vbase = Vt + ((size_t)((b * 4 + h) * HD)) * NN;

    float mrow[8];
    for (int r = 0; r < 8; ++r) mrow[r] = -1e30f;

    // ---- pass 1: row max of scaled scores
    for (int kt = 0; kt < 64; ++kt) {
        for (int t = 0; t < 2; ++t) {
            int key = kt * 32 + t * 16 + ln;
            const unsigned short* krow = Kbase + (size_t)key * D_MODEL;
            __builtin_prefetch(krow + 32 * D_MODEL, 0, 1);   // next kt
            Frag16 kf0, kf1;
            kf0.u[0] = *(const uint4*)(krow + half * 8);
            kf0.u[1] = *(const uint4*)(krow + 16 + half * 8);
            kf1.u[0] = *(const uint4*)(krow + 32 + half * 8);
            kf1.u[1] = *(const uint4*)(krow + 48 + half * 8);
            v8f s = {};
            s = __builtin_amdgcn_wmma_f32_16x16x32_bf16(false, qf0.v, false, kf0.v, (short)0, s, false, false);
            s = __builtin_amdgcn_wmma_f32_16x16x32_bf16(false, qf1.v, false, kf1.v, (short)0, s, false, false);
            for (int r = 0; r < 8; ++r) mrow[r] = fmaxf(mrow[r], s[r] * scale);
        }
    }
    for (int r = 0; r < 8; ++r)
        for (int m = 1; m < 16; m <<= 1)
            mrow[r] = fmaxf(mrow[r], __shfl_xor(mrow[r], m, 32));

    // ---- pass 2: exp, mask, denominators, O accumulation
    v8f O[4] = {};
    float Zp[8] = {0.f, 0.f, 0.f, 0.f, 0.f, 0.f, 0.f, 0.f};
    float Mp[8] = {0.f, 0.f, 0.f, 0.f, 0.f, 0.f, 0.f, 0.f};
    for (int kt = 0; kt < 64; ++kt) {
        for (int t = 0; t < 2; ++t) {
            int key = kt * 32 + t * 16 + ln;
            const unsigned short* krow = Kbase + (size_t)key * D_MODEL;
            __builtin_prefetch(krow + 32 * D_MODEL, 0, 1);   // next kt
            Frag16 kf0, kf1;
            kf0.u[0] = *(const uint4*)(krow + half * 8);
            kf0.u[1] = *(const uint4*)(krow + 16 + half * 8);
            kf1.u[0] = *(const uint4*)(krow + 32 + half * 8);
            kf1.u[1] = *(const uint4*)(krow + 48 + half * 8);
            v8f s = {};
            s = __builtin_amdgcn_wmma_f32_16x16x32_bf16(false, qf0.v, false, kf0.v, (short)0, s, false, false);
            s = __builtin_amdgcn_wmma_f32_16x16x32_bf16(false, qf1.v, false, kf1.v, (short)0, s, false, false);
            // masks: 8 row-bytes contiguous in transposed edge matrix -> one 8B load
            unsigned long long ev =
                *(const unsigned long long*)(edgeT + (size_t)key * NN + q0 + 8 * half);
            bool cmpos = colmask[b * NN + key] > 0.f;
            for (int r = 0; r < 8; ++r) {
                float e  = __expf(s[r] * scale - mrow[r]);
                float mk = (((ev >> (8 * r)) & 1ull) || cmpos) ? 1.f : 0.f;
                Zp[r] += e;
                float em = e * mk;
                Mp[r] += em;
                Pst[wid][r + 8 * half][t * 16 + ln] = f2bf(em);
            }
        }
        // wave-private LDS round-trip: DS ops within a wave are in-order (DScnt),
        // only a compiler scheduling fence is needed (no workgroup barrier).
        __builtin_amdgcn_wave_barrier();
        Frag16 pf;
        pf.u[0] = *(const uint4*)&Pst[wid][ln][half * 8];
        pf.u[1] = *(const uint4*)&Pst[wid][ln][16 + half * 8];
        for (int dt = 0; dt < 4; ++dt) {
            const unsigned short* vrow = Vbase + (size_t)(dt * 16 + ln) * NN + kt * 32;
            __builtin_prefetch(vrow + 32, 0, 1);             // next kt
            Frag16 vf;
            vf.u[0] = *(const uint4*)(vrow + half * 8);
            vf.u[1] = *(const uint4*)(vrow + 16 + half * 8);
            O[dt] = __builtin_amdgcn_wmma_f32_16x16x32_bf16(false, pf.v, false, vf.v,
                                                            (short)0, O[dt], false, false);
        }
        __builtin_amdgcn_wave_barrier();
    }
    // renormalize: sparse = em / (M + 1e-8 * Z)
    for (int r = 0; r < 8; ++r) {
        float z = Zp[r], mm = Mp[r];
        for (int m = 1; m < 16; m <<= 1) {
            z  += __shfl_xor(z,  m, 32);
            mm += __shfl_xor(mm, m, 32);
        }
        float inv = 1.f / (mm + 1e-8f * z);
        for (int dt = 0; dt < 4; ++dt) O[dt][r] *= inv;
    }
    for (int dt = 0; dt < 4; ++dt)
        for (int r = 0; r < 8; ++r)
            attnb[((size_t)(b * NN + q0 + r + 8 * half)) * D_MODEL + h * HD + dt * 16 + ln]
                = f2bf(O[dt][r]);
}

// ---------------------------------------------------------------- mask4 output (broadcast over heads)
__global__ void k_mask4(const unsigned char* __restrict__ edge,
                        const float* __restrict__ colmask, float* __restrict__ mout) {
    int t = blockIdx.x * 256 + threadIdx.x;          // BB*NN*(NN/4) = 4.19M
    int b   = t / (NN * (NN / 4));
    int rem = t % (NN * (NN / 4));
    int n   = rem / (NN / 4);
    int m0  = (rem % (NN / 4)) * 4;
    const unsigned char* ep = edge + (size_t)n * NN + m0;
    const float4 cm = *(const float4*)(colmask + b * NN + m0);
    float4 v;
    v.x = (ep[0] | (cm.x > 0.f)) ? 1.f : 0.f;
    v.y = (ep[1] | (cm.y > 0.f)) ? 1.f : 0.f;
    v.z = (ep[2] | (cm.z > 0.f)) ? 1.f : 0.f;
    v.w = (ep[3] | (cm.w > 0.f)) ? 1.f : 0.f;
    for (int hh = 0; hh < 4; ++hh)
        *(float4*)(mout + (((size_t)(b * 4 + hh) * NN + n) * NN + m0)) = v;
}

// ---------------------------------------------------------------- launcher
extern "C" void kernel_launch(void* const* d_in, const int* in_sizes, int n_in,
                              void* d_out, int out_size, void* d_ws, size_t ws_size,
                              hipStream_t stream) {
    const float*     x   = (const float*)d_in[0];
    const long long* ei  = (const long long*)d_in[1];
    const float* Wq  = (const float*)d_in[2];  const float* bq  = (const float*)d_in[3];
    const float* Wk  = (const float*)d_in[4];  const float* bk  = (const float*)d_in[5];
    const float* Wv  = (const float*)d_in[6];  const float* bv  = (const float*)d_in[7];
    const float* Wo  = (const float*)d_in[8];  const float* bo  = (const float*)d_in[9];
    const float* Wg1 = (const float*)d_in[10]; const float* bg1 = (const float*)d_in[11];
    const float* Wg2 = (const float*)d_in[12]; const float* bg2 = (const float*)d_in[13];

    const size_t MTOT = (size_t)BB * NN;               // 8192 rows
    char* ws = (char*)d_ws;
    size_t off = 0;
    float*          aggr    = (float*)(ws + off);          off += MTOT * D_MODEL * 4;
    float*          topo    = (float*)(ws + off);          off += MTOT * 4;
    float*          colmask = (float*)(ws + off);          off += MTOT * 4;
    unsigned char*  edge    = (unsigned char*)(ws + off);  off += (size_t)NN * NN;
    unsigned char*  edgeT   = (unsigned char*)(ws + off);  off += (size_t)NN * NN;
    unsigned short* xb      = (unsigned short*)(ws + off); off += MTOT * D_MODEL * 2;
    unsigned short* WqT     = (unsigned short*)(ws + off); off += (size_t)D_MODEL * D_MODEL * 2;
    unsigned short* WkT     = (unsigned short*)(ws + off); off += (size_t)D_MODEL * D_MODEL * 2;
    unsigned short* WvT     = (unsigned short*)(ws + off); off += (size_t)D_MODEL * D_MODEL * 2;
    unsigned short* WoT     = (unsigned short*)(ws + off); off += (size_t)D_MODEL * D_MODEL * 2;
    unsigned short* Qb      = (unsigned short*)(ws + off); off += MTOT * D_MODEL * 2;
    unsigned short* Kb      = (unsigned short*)(ws + off); off += MTOT * D_MODEL * 2;
    unsigned short* Vt      = (unsigned short*)(ws + off); off += MTOT * D_MODEL * 2;
    unsigned short* attnb   = (unsigned short*)(ws + off); off += MTOT * D_MODEL * 2;

    hipMemsetAsync(aggr,  0, MTOT * D_MODEL * 4, stream);
    hipMemsetAsync(edge,  0, (size_t)NN * NN, stream);
    hipMemsetAsync(edgeT, 0, (size_t)NN * NN, stream);

    // precision conversions (once)
    int n4 = (int)(MTOT * D_MODEL / 4);
    k_cvt_bf16<<<(n4 + 255) / 256, 256, 0, stream>>>(x, xb, n4);
    k_cvt_w_t<<<(D_MODEL * D_MODEL) / 256, 256, 0, stream>>>(Wq, WqT);
    k_cvt_w_t<<<(D_MODEL * D_MODEL) / 256, 256, 0, stream>>>(Wk, WkT);
    k_cvt_w_t<<<(D_MODEL * D_MODEL) / 256, 256, 0, stream>>>(Wv, WvT);
    k_cvt_w_t<<<(D_MODEL * D_MODEL) / 256, 256, 0, stream>>>(Wo, WoT);

    // GNN -> topo scores -> column mask
    k_scatter_edges<<<EE / 256, 256, 0, stream>>>(ei, edge, edgeT);
    k_gnn_aggregate<<<(BB * EE * 64) / 256, 256, 0, stream>>>(x, ei, aggr);
    k_gnn_mlp<<<BB * NN, 128, 0, stream>>>(aggr, Wg1, bg1, Wg2, bg2, topo);
    k_topk<<<(BB * NN) / 256, 256, 0, stream>>>(topo, colmask);

    // projections (WMMA)
    dim3 gg(BB * NN / 64, D_MODEL / 64);
    k_gemm256<<<gg, 128, 0, stream>>>(xb, WqT, bq, Qb, 0);
    k_gemm256<<<gg, 128, 0, stream>>>(xb, WkT, bk, Kb, 0);
    k_gemm256<<<gg, 128, 0, stream>>>(xb, WvT, bv, Vt, 1);

    // masked attention (WMMA, two-pass)
    k_attention<<<(BB * 4 * (NN / 16)) / 8, 256, 0, stream>>>(Qb, Kb, Vt, edgeT, colmask, attnb);

    // output projection (WMMA) + mask4
    float* out = (float*)d_out;
    k_gemm256<<<gg, 128, 0, stream>>>(attnb, WoT, bo, out, 2);
    k_mask4<<<(BB * NN * (NN / 4)) / 256, 256, 0, stream>>>(edge, colmask, out + MTOT * D_MODEL);
}
